// LSTM_38646115729591
// MI455X (gfx1250) — matrix operand order repaired and checked by reading, not compiled, not run on hardware
//
#include <hip/hip_runtime.h>
#include <hip/hip_bf16.h>

// ---------------------------------------------------------------------------
// Bidirectional 2-layer LSTM for MI455X (gfx1250, wave32, WMMA).
//   * bf16 WMMA (16x16x32) with f32 accumulation; f32 cell state in registers.
//   * 16 persistent WGs per direction; each owns a 32-wide hidden slice.
//   * W_hh slice (130 KB, bank-padded) staged ONCE in LDS; h_{t-1} (65 KB,
//     bank-padded) staged per step -> the serial h @ W_hh^T matmul runs
//     entirely out of LDS (ds_load_b128 fragments).
//   * Split arrive/wait device-scope barrier: the x-projection of step s+1
//     overlaps other WGs finishing step s; acquire-load polling.
//   * global_prefetch of next timestep's x tile; fast exp/rcp activations.
// ---------------------------------------------------------------------------

typedef __bf16 bf16;
typedef __attribute__((ext_vector_type(8)))  __bf16 v8bf;
typedef __attribute__((ext_vector_type(16))) __bf16 v16bf;
typedef __attribute__((ext_vector_type(8)))  float  v8f;

#define TT  512   // seq len
#define BB  64    // batch
#define DD  512   // layer-0 input dim
#define HH  512   // hidden
#define NWG 16    // workgroups per direction
#define HS  32    // hidden units per workgroup (HH / NWG)
#define LDP 520   // LDS row pitch in bf16 (512 + 8 pad -> conflict-free frags)

__device__ __forceinline__ v8f wmma_bf16(v16bf a, v16bf b, v8f c) {
  return __builtin_amdgcn_wmma_f32_16x16x32_bf16(false, a, false, b, (short)0, c,
                                                 false, false);
}

// A fragment (16x32 bf16) from global: lane = (m = lane&15, hi = lane>>4).
__device__ __forceinline__ v16bf load_afrag(const bf16* base, int ld, int mbase,
                                            int k, int nlo, int hi) {
  const bf16* p = base + (size_t)(mbase + nlo) * ld + k + hi * 8;
  v8bf lo  = *(const v8bf*)(p);
  v8bf hi8 = *(const v8bf*)(p + 16);
  return __builtin_shufflevector(lo, hi8, 0, 1, 2, 3, 4, 5, 6, 7,
                                 8, 9, 10, 11, 12, 13, 14, 15);
}

// A fragment from padded LDS tile.
__device__ __forceinline__ v16bf load_afrag_lds(const bf16* base, int mbase,
                                                int k, int nlo, int hi) {
  const bf16* p = base + (mbase + nlo) * LDP + k + hi * 8;
  v8bf lo  = *(const v8bf*)(p);
  v8bf hi8 = *(const v8bf*)(p + 16);
  return __builtin_shufflevector(lo, hi8, 0, 1, 2, 3, 4, 5, 6, 7,
                                 8, 9, 10, 11, 12, 13, 14, 15);
}

// B fragment (32x16 bf16) from weights stored [N, K] row-major (global).
__device__ __forceinline__ v16bf load_bfrag(const bf16* w, int ld, int nbase,
                                            int k, int nlo, int hi) {
  const bf16* p = w + (size_t)(nbase + nlo) * ld + k + hi * 8;
  v8bf lo = *(const v8bf*)(p);
  v8bf h8 = *(const v8bf*)(p + 16);
  return __builtin_shufflevector(lo, h8, 0, 1, 2, 3, 4, 5, 6, 7,
                                 8, 9, 10, 11, 12, 13, 14, 15);
}

// B fragment from padded LDS tile (16B-aligned halves -> ds_load_b128 x2).
__device__ __forceinline__ v16bf load_bfrag_lds(const bf16* w, int nbase,
                                                int k, int nlo, int hi) {
  const bf16* p = w + (nbase + nlo) * LDP + k + hi * 8;
  v8bf lo = *(const v8bf*)(p);
  v8bf h8 = *(const v8bf*)(p + 16);
  return __builtin_shufflevector(lo, h8, 0, 1, 2, 3, 4, 5, 6, 7,
                                 8, 9, 10, 11, 12, 13, 14, 15);
}

// Fast activations on the serial path: v_exp_f32 + v_rcp_f32.
__device__ __forceinline__ float fast_sigmoid(float x) {
  return __builtin_amdgcn_rcpf(1.0f + __expf(-x));
}
__device__ __forceinline__ float fast_tanh(float x) {
  return 1.0f - 2.0f * __builtin_amdgcn_rcpf(__expf(2.0f * x) + 1.0f);
}

template <int DIN, bool F32OUT>
__global__ __launch_bounds__(256, 1) void lstm_dir_kernel(
    const void* xin_,                                   // [T][B][DIN] bf16
    const void* wihf_, const void* whhf_, const float* biasf,
    const void* wihb_, const void* whhb_, const float* biasb,
    void* xout_,                                        // [T][B][2H] bf16 (L0)
    float* out_f32,                                     // [T][B][2H] f32  (L1)
    void* hbuf_,                                        // [2 dir][2 slot][B][H] bf16
    unsigned* bar)                                      // 2 counters (per dir)
{
  const bf16* xin = (const bf16*)xin_;
  bf16* xout = (bf16*)xout_;
  bf16* hbuf = (bf16*)hbuf_;

  const int dir  = blockIdx.y;        // 0 = fwd, 1 = bwd
  const int wg   = blockIdx.x;        // 0..NWG-1 -> hidden slice
  const int hb   = wg * HS;
  const int lane = threadIdx.x & 31;
  const int wv   = threadIdx.x >> 5;  // 8 waves
  const int mt   = wv >> 1;           // batch tile 0..3 (rows mt*16 .. +16)
  const int jt   = wv & 1;            // 16-col hidden tile in slice
  const int nlo  = lane & 15;
  const int hi   = lane >> 4;

  const bf16*  w_ih = (const bf16*)(dir ? wihb_ : wihf_);
  const bf16*  w_hh = (const bf16*)(dir ? whhb_ : whhf_);
  const float* bias = dir ? biasb : biasf;
  bf16* hping = hbuf + (size_t)dir * 2 * BB * HH;
  unsigned* ctr = bar + dir;

  __shared__ bf16 wlds[4 * HS * LDP];           // 130 KB: resident W_hh slice
  __shared__ bf16 hlds[BB * LDP];               //  65 KB: per-step h_{t-1}

  // ---- stage this WG's W_hh slice (4 gates x HS rows x 512 k) into LDS ----
  for (int idx = threadIdx.x; idx < 4 * HS * 32; idx += 256) {
    const int lr = idx >> 5;                    // 0..127 local row
    const int c  = (idx & 31) * 16;             // 16-elem chunk
    const int g  = lr >> 5;
    const int r  = lr & 31;
    const bf16* p = w_hh + (size_t)(g * HH + hb + r) * HH + c;
    v8bf v0 = *(const v8bf*)(p);
    v8bf v1 = *(const v8bf*)(p + 8);
    *(v8bf*)&wlds[lr * LDP + c]     = v0;
    *(v8bf*)&wlds[lr * LDP + c + 8] = v1;
  }
  __syncthreads();

  int   wrow[4];
  int   lrow[4];
  float bv[4];
#pragma unroll
  for (int g = 0; g < 4; ++g) {                 // PyTorch gate order i,f,g,o
    wrow[g] = g * HH + hb + jt * 16;
    lrow[g] = g * HS + jt * 16;
    bv[g]   = bias[wrow[g] + nlo];
  }

  float cst[8];                                 // persistent cell state (f32)
#pragma unroll
  for (int e = 0; e < 8; ++e) cst[e] = 0.f;

  for (int s = 0; s < TT; ++s) {
    const int t = dir ? (TT - 1 - s) : s;

    // prefetch next timestep's x tile while peers finish step s-1
    if (s + 1 < TT) {
      const int tn = dir ? (TT - 2 - s) : (s + 1);
      const char* nx = (const char*)(xin + (size_t)tn * BB * DIN);
      for (int o = threadIdx.x * 256; o < BB * DIN * 2; o += 256 * 256)
        __builtin_prefetch(nx + o, 0, 0);
    }

    v8f acc[4];
#pragma unroll
    for (int g = 0; g < 4; ++g) {
      v8f c0;
#pragma unroll
      for (int e = 0; e < 8; ++e) c0[e] = bv[g];
      acc[g] = c0;
    }

    // gates += x_t @ W_ih^T   (no dependence on h: overlaps peers' step s-1)
    const bf16* xrow = xin + (size_t)t * BB * DIN;
    for (int k = 0; k < DIN; k += 32) {
      v16bf a = load_afrag(xrow, DIN, mt * 16, k, nlo, hi);
#pragma unroll
      for (int g = 0; g < 4; ++g) {
        v16bf wb = load_bfrag(w_ih, DIN, wrow[g], k, nlo, hi);
        acc[g] = wmma_bf16(a, wb, acc[g]);
      }
    }

    // gates += h_{s-1} @ W_hh^T  (both operands from LDS; skip at s==0)
    if (s > 0) {
      if (threadIdx.x == 0) {                   // barrier WAIT (acquire poll)
        const unsigned target = (unsigned)s * NWG;
        while (__hip_atomic_load(ctr, __ATOMIC_ACQUIRE,
                                 __HIP_MEMORY_SCOPE_AGENT) < target)
          __builtin_amdgcn_s_sleep(1);
      }
      __syncthreads();
      __threadfence();

      // cooperative stage of h_{s-1} (64x512 bf16) into bank-padded LDS
      const bf16* hprev = hping + (size_t)((s & 1) ^ 1) * BB * HH;
      for (int idx = threadIdx.x; idx < BB * 32; idx += 256) {
        const int r = idx >> 5;
        const int c = (idx & 31) * 16;
        const bf16* p = hprev + r * HH + c;
        v8bf v0 = *(const v8bf*)(p);
        v8bf v1 = *(const v8bf*)(p + 8);
        *(v8bf*)&hlds[r * LDP + c]     = v0;
        *(v8bf*)&hlds[r * LDP + c + 8] = v1;
      }
      __syncthreads();

      for (int k = 0; k < HH; k += 32) {
        v16bf a = load_afrag_lds(hlds, mt * 16, k, nlo, hi);
#pragma unroll
        for (int g = 0; g < 4; ++g) {
          v16bf wb = load_bfrag_lds(wlds, lrow[g], k, nlo, hi);
          acc[g] = wmma_bf16(a, wb, acc[g]);
        }
      }
      __syncthreads();    // hlds reuse guard for next step
    }

    // pointwise cell update + writes (C/D layout: VGPR r -> M=r+8*hi, N=lane&15)
    bf16* hwr = hping + (size_t)(s & 1) * BB * HH;
    const int hcol = hb + jt * 16 + nlo;
    {
      v8f iG = acc[0], fG = acc[1], gG = acc[2], oG = acc[3];
#pragma unroll
      for (int e = 0; e < 8; ++e) {
        const int batch = mt * 16 + e + hi * 8;
        float c = fast_sigmoid(fG[e]) * cst[e] +
                  fast_sigmoid(iG[e]) * fast_tanh(gG[e]);
        float h = fast_sigmoid(oG[e]) * fast_tanh(c);
        cst[e] = c;
        hwr[(size_t)batch * HH + hcol] = (bf16)h;
        const size_t oidx =
            ((size_t)t * BB + batch) * (2 * HH) + (size_t)dir * HH + hcol;
        if (F32OUT) out_f32[oidx] = h;
        else        xout[oidx]    = (bf16)h;
      }
    }

    // barrier ARRIVE phase (release h_s)
    __threadfence();
    __syncthreads();
    if (threadIdx.x == 0)
      __hip_atomic_fetch_add(ctr, 1u, __ATOMIC_RELEASE,
                             __HIP_MEMORY_SCOPE_AGENT);
  }
}

// ---------------- prep kernels ----------------

__global__ void cvt_f32_bf16(const float* __restrict__ s, void* __restrict__ d_,
                             long long n) {
  bf16* d = (bf16*)d_;
  long long i = (long long)blockIdx.x * blockDim.x + threadIdx.x;
  if (i < n) d[i] = (bf16)s[i];
}

__global__ void bias_sum(const float* __restrict__ a, const float* __restrict__ b,
                         float* __restrict__ o, int n) {
  int i = blockIdx.x * blockDim.x + threadIdx.x;
  if (i < n) o[i] = a[i] + b[i];
}

__global__ void init_bar(unsigned* bar) {
  if (threadIdx.x < 8) bar[threadIdx.x] = 0u;
}

// ---------------- host launcher ----------------

extern "C" void kernel_launch(void* const* d_in, const int* in_sizes, int n_in,
                              void* d_out, int out_size, void* d_ws, size_t ws_size,
                              hipStream_t stream) {
  (void)in_sizes; (void)n_in; (void)out_size; (void)ws_size;

  const float* x     = (const float*)d_in[0];
  const float* wih0f = (const float*)d_in[1];
  const float* whh0f = (const float*)d_in[2];
  const float* bih0f = (const float*)d_in[3];
  const float* bhh0f = (const float*)d_in[4];
  const float* wih0b = (const float*)d_in[5];
  const float* whh0b = (const float*)d_in[6];
  const float* bih0b = (const float*)d_in[7];
  const float* bhh0b = (const float*)d_in[8];
  const float* wih1f = (const float*)d_in[9];
  const float* whh1f = (const float*)d_in[10];
  const float* bih1f = (const float*)d_in[11];
  const float* bhh1f = (const float*)d_in[12];
  const float* wih1b = (const float*)d_in[13];
  const float* whh1b = (const float*)d_in[14];
  const float* bih1b = (const float*)d_in[15];
  const float* bhh1b = (const float*)d_in[16];

  char*  ws  = (char*)d_ws;
  size_t off = 0;
  auto alloc = [&](size_t bytes) -> void* {
    void* p = ws + off;
    off += (bytes + 255) & ~(size_t)255;
    return p;
  };

  unsigned* bar = (unsigned*)alloc(8 * sizeof(unsigned));
  void* xbf    = alloc((size_t)TT * BB * DD * 2);        // x in bf16
  void* x1bf   = alloc((size_t)TT * BB * 2 * HH * 2);    // layer-0 output, bf16
  void* bwih0f = alloc((size_t)2048 * 512 * 2);
  void* bwhh0f = alloc((size_t)2048 * 512 * 2);
  void* bwih0b = alloc((size_t)2048 * 512 * 2);
  void* bwhh0b = alloc((size_t)2048 * 512 * 2);
  void* bwih1f = alloc((size_t)2048 * 1024 * 2);
  void* bwhh1f = alloc((size_t)2048 * 512 * 2);
  void* bwih1b = alloc((size_t)2048 * 1024 * 2);
  void* bwhh1b = alloc((size_t)2048 * 512 * 2);
  float* bs0f  = (float*)alloc(2048 * sizeof(float));
  float* bs0b  = (float*)alloc(2048 * sizeof(float));
  float* bs1f  = (float*)alloc(2048 * sizeof(float));
  float* bs1b  = (float*)alloc(2048 * sizeof(float));
  void* hbuf   = alloc((size_t)2 * 2 * BB * HH * 2);     // h ping-pong, bf16

  dim3 blk(256);
  init_bar<<<1, 32, 0, stream>>>(bar);

  auto cvt = [&](const float* s, void* d, long long n) {
    cvt_f32_bf16<<<(unsigned)((n + 255) / 256), blk, 0, stream>>>(s, d, n);
  };
  cvt(x,     xbf,    (long long)TT * BB * DD);
  cvt(wih0f, bwih0f, 2048LL * 512);
  cvt(whh0f, bwhh0f, 2048LL * 512);
  cvt(wih0b, bwih0b, 2048LL * 512);
  cvt(whh0b, bwhh0b, 2048LL * 512);
  cvt(wih1f, bwih1f, 2048LL * 1024);
  cvt(whh1f, bwhh1f, 2048LL * 512);
  cvt(wih1b, bwih1b, 2048LL * 1024);
  cvt(whh1b, bwhh1b, 2048LL * 512);

  bias_sum<<<8, blk, 0, stream>>>(bih0f, bhh0f, bs0f, 2048);
  bias_sum<<<8, blk, 0, stream>>>(bih0b, bhh0b, bs0b, 2048);
  bias_sum<<<8, blk, 0, stream>>>(bih1f, bhh1f, bs1f, 2048);
  bias_sum<<<8, blk, 0, stream>>>(bih1b, bhh1b, bs1b, 2048);

  // Layer 0: input = x (bf16), output = x1bf (bf16)
  lstm_dir_kernel<DD, false><<<dim3(NWG, 2), blk, 0, stream>>>(
      xbf, bwih0f, bwhh0f, bs0f, bwih0b, bwhh0b, bs0b,
      x1bf, nullptr, hbuf, bar + 0);

  // Layer 1: input = x1bf (bf16), output = d_out (f32)
  lstm_dir_kernel<2 * HH, true><<<dim3(NWG, 2), blk, 0, stream>>>(
      x1bf, bwih1f, bwhh1f, bs1f, bwih1b, bwhh1b, bs1b,
      nullptr, (float*)d_out, hbuf, bar + 2);
}
